// SimplePeriodicNetwork_5334349381938
// MI455X (gfx1250) — compile-verified, matrix operand order checked by reading, not run on hardware
//
#include <hip/hip_runtime.h>
#include <math.h>

// ---------------------------------------------------------------------------
// e3nn-style periodic network forward, CDNA5 (gfx1250, wave32, WMMA).
//
// Pipeline per layer:
//   node_lin1_sc : xbuf = lin1(feat), scb = sc(feat)          (VALU, mem-bound)
//   conv_edge    : fused edge MLP (WMMA f16->f32) + tensor    (WMMA + LDS + atomics)
//                  product + scatter-add into mid[dst]
//   node_out     : lin2(mid), skip-blend, gate -> next feat   (VALU)
// ---------------------------------------------------------------------------

typedef __attribute__((ext_vector_type(16))) _Float16 v16h;
typedef __attribute__((ext_vector_type(8)))  float    v8f;

#define RS3   0.5773502691896258f    // 1/sqrt(3)
#define RS6   0.4082482904638631f    // 1/sqrt(6)
#define RS10  0.31622776601683794f   // 1/sqrt(10)  (NBASIS and NUM_NEIGHBORS)
#define SQ3F  1.7320508075688772f
#define C_Sf  0.3826834323650898f    // sin(pi/8)
#define C_Xf  0.9238795325112867f    // cos(pi/8)

// silu via hardware v_rcp_f32 (avoids IEEE div_scale/div_fmas expansion)
__device__ __forceinline__ float silu_(float v) {
  return v * __builtin_amdgcn_rcpf(1.0f + __expf(-v));
}

// K-index held by element i of a v16h A/B fragment for lane-group g (wave32,
// f16 WMMA 16x16x32 layout: lanes 0-15 rows, VGPR0..3 K=0..7 (g=0)/8..15 (g=1),
// VGPR4..7 K=16..23 / 24..31).
__device__ __forceinline__ int frag_k(int g, int i) {
  return ((i >> 3) << 4) + (g << 3) + (i & 7);
}

// ---------------------------------------------------------------------------
// Edge geometry: vec, r, spherical harmonics (l=1), radial cosine basis.
// ---------------------------------------------------------------------------
__global__ void edge_geom(const float* __restrict__ pos, const int* __restrict__ eidx,
                          const float* __restrict__ eshift, const float* __restrict__ lattice,
                          const int* __restrict__ batch,
                          float* __restrict__ sh, float* __restrict__ emb, int E) {
  int e = blockIdx.x * blockDim.x + threadIdx.x;
  if (e >= E) return;
  int s = eidx[e], d = eidx[E + e];
  const float* L = lattice + (size_t)batch[s] * 9;
  float sx = eshift[e * 3 + 0], sy = eshift[e * 3 + 1], sz = eshift[e * 3 + 2];
  float v0 = pos[d * 3 + 0] - pos[s * 3 + 0] + sx * L[0] + sy * L[3] + sz * L[6];
  float v1 = pos[d * 3 + 1] - pos[s * 3 + 1] + sx * L[1] + sy * L[4] + sz * L[7];
  float v2 = pos[d * 3 + 2] - pos[s * 3 + 2] + sx * L[2] + sy * L[5] + sz * L[8];
  float r = sqrtf(v0 * v0 + v1 * v1 + v2 * v2 + 1e-12f);
  float inv = __builtin_amdgcn_rcpf(r);
  // sh '1o' = sqrt(3) * (u_y, u_z, u_x)
  sh[e * 3 + 0] = SQ3F * v1 * inv;
  sh[e * 3 + 1] = SQ3F * v2 * inv;
  sh[e * 3 + 2] = SQ3F * v0 * inv;
  const float step = 5.0f / 11.0f;
  const float rstep = 11.0f / 5.0f;
#pragma unroll
  for (int j = 0; j < 10; ++j) {
    float dd = (r - step * (float)(j + 1)) * rstep;
    float msk = (dd > -1.0f && dd < 1.0f) ? 1.0f : 0.0f;
    emb[e * 10 + j] = __cosf(1.5707963267948966f * dd) * msk * 3.1622776601683795f;
  }
}

// ---------------------------------------------------------------------------
// Per-node lin1 (-> xbuf, edge-gather source) and sc skip path (-> scb).
// All input multiplicities are 16 => scale 1/sqrt(16) = 0.25.
// ---------------------------------------------------------------------------
template <int LAYER>
__global__ void node_lin1_sc(const float* __restrict__ feat,
                             const float* __restrict__ w0e, const float* __restrict__ w1o,
                             const float* __restrict__ w1e,
                             const float* __restrict__ sc0e, const float* __restrict__ sc1o,
                             float* __restrict__ xbuf, float* __restrict__ scb, int N) {
  int n = blockIdx.x * blockDim.x + threadIdx.x;
  if (n >= N) return;
  const float Q = 0.25f;
  if constexpr (LAYER == 0) {
    const float* f = feat + (size_t)n * 16;
    float* xo = xbuf + (size_t)n * 16;
    float* so = scb + (size_t)n * 32;
    for (int v = 0; v < 16; ++v) { float a = 0; for (int u = 0; u < 16; ++u) a += f[u] * w0e[u * 16 + v]; xo[v] = a * Q; }
    for (int v = 0; v < 32; ++v) { float a = 0; for (int u = 0; u < 16; ++u) a += f[u] * sc0e[u * 32 + v]; so[v] = a * Q; }
  } else if constexpr (LAYER == 1) {
    const float* f = feat + (size_t)n * 64;   // [0e:16][1o:16x3]
    float* xo = xbuf + (size_t)n * 64;
    float* so = scb + (size_t)n * 96;         // [sc0e:48][sc1o:16x3]
    for (int v = 0; v < 16; ++v) { float a = 0; for (int u = 0; u < 16; ++u) a += f[u] * w0e[u * 16 + v]; xo[v] = a * Q; }
    for (int v = 0; v < 16; ++v)
      for (int m = 0; m < 3; ++m) { float a = 0; for (int u = 0; u < 16; ++u) a += f[16 + u * 3 + m] * w1o[u * 16 + v]; xo[16 + v * 3 + m] = a * Q; }
    for (int v = 0; v < 48; ++v) { float a = 0; for (int u = 0; u < 16; ++u) a += f[u] * sc0e[u * 48 + v]; so[v] = a * Q; }
    for (int v = 0; v < 16; ++v)
      for (int m = 0; m < 3; ++m) { float a = 0; for (int u = 0; u < 16; ++u) a += f[16 + u * 3 + m] * sc1o[u * 16 + v]; so[48 + v * 3 + m] = a * Q; }
  } else {
    const float* f = feat + (size_t)n * 112;  // [0e:16][1o:16x3][1e:16x3]
    float* xo = xbuf + (size_t)n * 112;
    float* so = scb + (size_t)n * 3;
    for (int v = 0; v < 16; ++v) { float a = 0; for (int u = 0; u < 16; ++u) a += f[u] * w0e[u * 16 + v]; xo[v] = a * Q; }
    for (int v = 0; v < 16; ++v)
      for (int m = 0; m < 3; ++m) { float a = 0; for (int u = 0; u < 16; ++u) a += f[16 + u * 3 + m] * w1o[u * 16 + v]; xo[16 + v * 3 + m] = a * Q; }
    for (int v = 0; v < 16; ++v)
      for (int m = 0; m < 3; ++m) { float a = 0; for (int u = 0; u < 16; ++u) a += f[64 + u * 3 + m] * w1e[u * 16 + v]; xo[64 + v * 3 + m] = a * Q; }
    for (int m = 0; m < 3; ++m) { float a = 0; for (int u = 0; u < 16; ++u) a += f[16 + u * 3 + m] * sc1o[u]; so[m] = a * Q; }
  }
}

// ---------------------------------------------------------------------------
// Fused edge kernel: one wave per 16-edge tile.
//   GEMM1 (WMMA f16): emb[16x10 pad 32] x W1[10x64] -> silu -> h (LDS, f16)
//   GEMM2 (WMMA f16): h[16x64] x W2[64xNUMEL]       -> w_edge (LDS, f32)
//   TP+scatter: lanes 0..31 = (edge = lane&15, channel-half = lane>>4)
// ---------------------------------------------------------------------------
template <int LAYER, int NUMEL, int XDIM, int MIDDIM>
__global__ void conv_edge(const int* __restrict__ eidx, const float* __restrict__ sh,
                          const float* __restrict__ emb, const float* __restrict__ xbuf,
                          const float* __restrict__ W1, const float* __restrict__ W2,
                          float* __restrict__ mid, int E) {
  constexpr int NT2 = NUMEL / 16;           // numel is always a multiple of 16
  __shared__ _Float16 hlds[4][16 * 64];
  __shared__ float    wlds[4][16 * NUMEL];

  const int wave = threadIdx.x >> 5;
  const int lane = threadIdx.x & 31;
  const int base = (blockIdx.x * 4 + wave) * 16;
  if (base >= E) return;                    // wave-uniform (E % 16 == 0)

  const int row = lane & 15;                // edge-in-tile / matrix row / column
  const int g   = lane >> 4;                // lane group
  const int e   = base + row;

  // ---- GEMM1: A = radial embedding (K=10, zero-padded to 32) ----
  v16h a{};
#pragma unroll
  for (int i = 0; i < 16; ++i) {
    int k = frag_k(g, i);
    a[i] = (k < 10) ? (_Float16)emb[(size_t)e * 10 + k] : (_Float16)0.0f;
  }
  _Float16* hrow = &hlds[wave][0];
#pragma unroll
  for (int t = 0; t < 4; ++t) {
    const int col = t * 16 + row;
    v16h b{};
#pragma unroll
    for (int i = 0; i < 16; ++i) {
      int k = frag_k(g, i);
      b[i] = (k < 10) ? (_Float16)W1[k * 64 + col] : (_Float16)0.0f;
    }
    v8f c{};
    c = __builtin_amdgcn_wmma_f32_16x16x32_f16(false, a, false, b, (short)0, c, false, false);
#pragma unroll
    for (int j = 0; j < 8; ++j) {
      int m = j + 8 * g;
      hrow[m * 64 + col] = (_Float16)silu_(c[j] * RS10);   // h = silu(emb@W1/sqrt(10))
    }
  }
  asm volatile("s_wait_dscnt 0x0" ::: "memory");           // cross-lane LDS visibility

  // ---- GEMM2: w_edge = h @ W2 / sqrt(64) ----
#pragma unroll
  for (int t = 0; t < NT2; ++t) {
    const int col = t * 16 + row;
    v8f c{};
#pragma unroll
    for (int kc = 0; kc < 2; ++kc) {
      v16h a2, b2;
#pragma unroll
      for (int i = 0; i < 16; ++i) {
        int kk = kc * 32 + frag_k(g, i);
        a2[i] = hrow[row * 64 + kk];
        b2[i] = (_Float16)W2[kk * NUMEL + col];
      }
      c = __builtin_amdgcn_wmma_f32_16x16x32_f16(false, a2, false, b2, (short)0, c, false, false);
    }
#pragma unroll
    for (int j = 0; j < 8; ++j) {
      int m = j + 8 * g;
      wlds[wave][m * NUMEL + col] = c[j] * 0.125f;
    }
  }
  asm volatile("s_wait_dscnt 0x0" ::: "memory");

  // ---- Tensor product + scatter (each lane: 8 channels of one edge) ----
  const int src = eidx[e];
  const int dst = eidx[E + e];
  const float s0 = sh[(size_t)e * 3 + 0], s1 = sh[(size_t)e * 3 + 1], s2 = sh[(size_t)e * 3 + 2];
  const float* xe = &xbuf[(size_t)src * XDIM];
  float* md = &mid[(size_t)dst * MIDDIM];
  const float* w = &wlds[wave][row * NUMEL];

#pragma unroll
  for (int uu = 0; uu < 8; ++uu) {
    const int u = g * 8 + uu;
    if constexpr (LAYER == 0) {
      // instrs: (0e,0e,0e)@0  (0e,1o,1o)@16 ; mid = [0e:16][1o:16x3]
      float x0 = xe[u];
      atomicAdd(&md[u], x0 * w[u] * RS10);
      float c2 = x0 * w[16 + u] * (RS3 * RS10);
      atomicAdd(&md[16 + u * 3 + 0], c2 * s0);
      atomicAdd(&md[16 + u * 3 + 1], c2 * s1);
      atomicAdd(&md[16 + u * 3 + 2], c2 * s2);
    } else if constexpr (LAYER == 1) {
      // instrs: (0e,0e,0e)@0 (0e,1o,1o)@16 (1o,0e,1o)@32 (1o,1o,0e)@48 (1o,1o,1e)@64
      // mid = [0e:32][1o:32x3 @32][1e:16x3 @128]
      float x0 = xe[u];
      float p0 = xe[16 + u * 3 + 0], p1 = xe[16 + u * 3 + 1], p2 = xe[16 + u * 3 + 2];
      atomicAdd(&md[u], x0 * w[u] * RS10);
      float c2 = x0 * w[16 + u] * (RS3 * RS10);
      atomicAdd(&md[32 + u * 3 + 0], c2 * s0);
      atomicAdd(&md[32 + u * 3 + 1], c2 * s1);
      atomicAdd(&md[32 + u * 3 + 2], c2 * s2);
      float c3 = w[32 + u] * (RS3 * RS10);
      atomicAdd(&md[32 + (16 + u) * 3 + 0], c3 * p0);
      atomicAdd(&md[32 + (16 + u) * 3 + 1], c3 * p1);
      atomicAdd(&md[32 + (16 + u) * 3 + 2], c3 * p2);
      atomicAdd(&md[16 + u], (p0 * s0 + p1 * s1 + p2 * s2) * w[48 + u] * (RS3 * RS10));
      float c5 = w[64 + u] * (RS6 * RS10);
      atomicAdd(&md[128 + u * 3 + 0], (p1 * s2 - p2 * s1) * c5);
      atomicAdd(&md[128 + u * 3 + 1], (p2 * s0 - p0 * s2) * c5);
      atomicAdd(&md[128 + u * 3 + 2], (p0 * s1 - p1 * s0) * c5);
    } else {
      // instrs: (0e,1o,1o)@0 (1o,0e,1o)@16 (1e,1o,1o)@32 ; mid = [1o:48x3]
      float x0 = xe[u];
      float p0 = xe[16 + u * 3 + 0], p1 = xe[16 + u * 3 + 1], p2 = xe[16 + u * 3 + 2];
      float q0 = xe[64 + u * 3 + 0], q1 = xe[64 + u * 3 + 1], q2 = xe[64 + u * 3 + 2];
      float c1 = x0 * w[u] * (RS3 * RS10);
      atomicAdd(&md[u * 3 + 0], c1 * s0);
      atomicAdd(&md[u * 3 + 1], c1 * s1);
      atomicAdd(&md[u * 3 + 2], c1 * s2);
      float c2 = w[16 + u] * (RS3 * RS10);
      atomicAdd(&md[(16 + u) * 3 + 0], c2 * p0);
      atomicAdd(&md[(16 + u) * 3 + 1], c2 * p1);
      atomicAdd(&md[(16 + u) * 3 + 2], c2 * p2);
      float c3 = w[32 + u] * (RS6 * RS10);
      atomicAdd(&md[(32 + u) * 3 + 0], (q1 * s2 - q2 * s1) * c3);
      atomicAdd(&md[(32 + u) * 3 + 1], (q2 * s0 - q0 * s2) * c3);
      atomicAdd(&md[(32 + u) * 3 + 2], (q0 * s1 - q1 * s0) * c3);
    }
  }
}

// ---------------------------------------------------------------------------
// lin2(mid) + skip blend + gate -> next features (or final output).
// ---------------------------------------------------------------------------
template <int LAYER>
__global__ void node_out(const float* __restrict__ mid, const float* __restrict__ scb,
                         const float* __restrict__ l2_0e, const float* __restrict__ l2_1o,
                         const float* __restrict__ l2_1e,
                         float* __restrict__ featOut, float* __restrict__ out, int N) {
  int n = blockIdx.x * blockDim.x + threadIdx.x;
  if (n >= N) return;
  if constexpr (LAYER == 0) {
    const float* md = mid + (size_t)n * 64;
    const float* sc = scb + (size_t)n * 32;
    float* fo = featOut + (size_t)n * 64;
    float gct[16];
    for (int v = 0; v < 32; ++v) {
      float a = 0; for (int u = 0; u < 16; ++u) a += md[u] * l2_0e[u * 32 + v];
      float y = C_Sf * sc[v] + C_Xf * a * 0.25f;
      if (v < 16) fo[v] = silu_(y); else gct[v - 16] = silu_(y);
    }
    for (int v = 0; v < 16; ++v)
      for (int m = 0; m < 3; ++m) {
        float a = 0; for (int u = 0; u < 16; ++u) a += md[16 + u * 3 + m] * l2_1o[u * 16 + v];
        fo[16 + v * 3 + m] = (a * 0.25f) * gct[v];          // '1o' has no sc skip here
      }
  } else if constexpr (LAYER == 1) {
    const float* md = mid + (size_t)n * 176;
    const float* sc = scb + (size_t)n * 96;
    float* fo = featOut + (size_t)n * 112;
    const float Q32 = 0.17677669529663687f;                 // 1/sqrt(32)
    float gct[32];
    for (int v = 0; v < 48; ++v) {
      float a = 0; for (int u = 0; u < 32; ++u) a += md[u] * l2_0e[u * 48 + v];
      float y = C_Sf * sc[v] + C_Xf * a * Q32;
      if (v < 16) fo[v] = silu_(y); else gct[v - 16] = silu_(y);
    }
    for (int v = 0; v < 16; ++v)
      for (int m = 0; m < 3; ++m) {
        float a = 0; for (int u = 0; u < 32; ++u) a += md[32 + u * 3 + m] * l2_1o[u * 16 + v];
        float y = C_Sf * sc[48 + v * 3 + m] + C_Xf * a * Q32;
        fo[16 + v * 3 + m] = y * gct[v];
      }
    for (int v = 0; v < 16; ++v)
      for (int m = 0; m < 3; ++m) {
        float a = 0; for (int u = 0; u < 16; ++u) a += md[128 + u * 3 + m] * l2_1e[u * 16 + v];
        fo[64 + v * 3 + m] = (a * 0.25f) * gct[16 + v];     // '1e' has no sc skip
      }
  } else {
    const float* md = mid + (size_t)n * 144;
    const float* sc = scb + (size_t)n * 3;
    const float Q48 = 0.14433756729740643f;                 // 1/sqrt(48)
    for (int m = 0; m < 3; ++m) {
      float a = 0; for (int u = 0; u < 48; ++u) a += md[u * 3 + m] * l2_1o[u];
      out[n * 3 + m] = C_Sf * sc[m] + C_Xf * a * Q48;
    }
  }
}

// ---------------------------------------------------------------------------
// Host orchestration. Params arrive pytree-flattened (list order, dict keys
// sorted: fc_w1, fc_w2, lin1{...}, lin2{...}, sc{...}).
// ---------------------------------------------------------------------------
extern "C" void kernel_launch(void* const* d_in, const int* in_sizes, int n_in,
                              void* d_out, int out_size, void* d_ws, size_t ws_size,
                              hipStream_t stream) {
  (void)n_in; (void)out_size; (void)ws_size;
  const float* pos     = (const float*)d_in[0];
  const float* xin     = (const float*)d_in[1];
  const int*   eidx    = (const int*)d_in[2];
  const float* eshift  = (const float*)d_in[3];
  const float* lattice = (const float*)d_in[4];
  const int*   batch   = (const int*)d_in[5];
  // layer 0: fc_w1, fc_w2, lin1[0e], lin2[0e], lin2[1o], sc[0e]
  const float* l0_fc1 = (const float*)d_in[6];
  const float* l0_fc2 = (const float*)d_in[7];
  const float* l0_lin1_0e = (const float*)d_in[8];
  const float* l0_lin2_0e = (const float*)d_in[9];
  const float* l0_lin2_1o = (const float*)d_in[10];
  const float* l0_sc_0e   = (const float*)d_in[11];
  // layer 1: fc_w1, fc_w2, lin1[0e,1o], lin2[0e,1e,1o], sc[0e,1o]
  const float* l1_fc1 = (const float*)d_in[12];
  const float* l1_fc2 = (const float*)d_in[13];
  const float* l1_lin1_0e = (const float*)d_in[14];
  const float* l1_lin1_1o = (const float*)d_in[15];
  const float* l1_lin2_0e = (const float*)d_in[16];
  const float* l1_lin2_1e = (const float*)d_in[17];
  const float* l1_lin2_1o = (const float*)d_in[18];
  const float* l1_sc_0e   = (const float*)d_in[19];
  const float* l1_sc_1o   = (const float*)d_in[20];
  // layer 2: fc_w1, fc_w2, lin1[0e,1e,1o], lin2[1o], sc[1o]
  const float* l2_fc1 = (const float*)d_in[21];
  const float* l2_fc2 = (const float*)d_in[22];
  const float* l2_lin1_0e = (const float*)d_in[23];
  const float* l2_lin1_1e = (const float*)d_in[24];
  const float* l2_lin1_1o = (const float*)d_in[25];
  const float* l2_lin2_1o = (const float*)d_in[26];
  const float* l2_sc_1o   = (const float*)d_in[27];

  const int N = in_sizes[0] / 3;
  const int E = in_sizes[2] / 2;

  char* ws = (char*)d_ws;
  size_t off = 0;
  auto carve = [&](size_t bytes) -> float* {
    float* p = (float*)(ws + off);
    off = (off + bytes + 255) & ~(size_t)255;
    return p;
  };
  float* sh    = carve((size_t)E * 3 * 4);     //   6.0 MB
  float* emb   = carve((size_t)E * 10 * 4);    //  20.0 MB
  float* xbuf  = carve((size_t)N * 112 * 4);   //  22.4 MB (lin1 features, edge-gather src)
  float* scb   = carve((size_t)N * 96 * 4);    //  19.2 MB (skip path)
  float* mid   = carve((size_t)N * 176 * 4);   //  35.2 MB (scatter accumulator, L2-resident)
  float* featA = carve((size_t)N * 64 * 4);    //  12.8 MB
  float* featB = carve((size_t)N * 112 * 4);   //  22.4 MB

  edge_geom<<<(E + 255) / 256, 256, 0, stream>>>(pos, eidx, eshift, lattice, batch, sh, emb, E);

  const int nblk = (N + 127) / 128;
  const int eblk = ((E + 15) / 16 + 3) / 4;    // 4 waves (tiles) per 128-thread block

  // ---- layer 0 ----
  hipMemsetAsync(mid, 0, (size_t)N * 64 * 4, stream);
  node_lin1_sc<0><<<nblk, 128, 0, stream>>>(xin, l0_lin1_0e, nullptr, nullptr, l0_sc_0e, nullptr, xbuf, scb, N);
  conv_edge<0, 32, 16, 64><<<eblk, 128, 0, stream>>>(eidx, sh, emb, xbuf, l0_fc1, l0_fc2, mid, E);
  node_out<0><<<nblk, 128, 0, stream>>>(mid, scb, l0_lin2_0e, l0_lin2_1o, nullptr, featA, nullptr, N);

  // ---- layer 1 ----
  hipMemsetAsync(mid, 0, (size_t)N * 176 * 4, stream);
  node_lin1_sc<1><<<nblk, 128, 0, stream>>>(featA, l1_lin1_0e, l1_lin1_1o, nullptr, l1_sc_0e, l1_sc_1o, xbuf, scb, N);
  conv_edge<1, 80, 64, 176><<<eblk, 128, 0, stream>>>(eidx, sh, emb, xbuf, l1_fc1, l1_fc2, mid, E);
  node_out<1><<<nblk, 128, 0, stream>>>(mid, scb, l1_lin2_0e, l1_lin2_1o, l1_lin2_1e, featB, nullptr, N);

  // ---- layer 2 (final) ----
  hipMemsetAsync(mid, 0, (size_t)N * 144 * 4, stream);
  node_lin1_sc<2><<<nblk, 128, 0, stream>>>(featB, l2_lin1_0e, l2_lin1_1o, l2_lin1_1e, nullptr, l2_sc_1o, xbuf, scb, N);
  conv_edge<2, 48, 112, 144><<<eblk, 128, 0, stream>>>(eidx, sh, emb, xbuf, l2_fc1, l2_fc2, mid, E);
  node_out<2><<<nblk, 128, 0, stream>>>(mid, scb, nullptr, l2_lin2_1o, nullptr, nullptr, (float*)d_out, N);
}
